// AttentionClustering_13426067768099
// MI455X (gfx1250) — matrix operand order repaired and checked by reading, not compiled, not run on hardware
//
#include <hip/hip_runtime.h>
#include <hip/hip_bf16.h>

#define HH 256
#define WW 256
#define QQ 256

typedef __attribute__((ext_vector_type(16))) _Float16 v16h;
typedef __attribute__((ext_vector_type(8)))  _Float16 v8h;
typedef __attribute__((ext_vector_type(8)))  float    v8f;

// Build a 16x32 f16 A-fragment (one lane's 16 halves) from channel-contiguous data.
// Lanes 0-15 hold K=0..15, lanes 16-31 hold K=16..31 (g = lane>=16).
// Works for both global and LDS (address-space preserved => ds_load_b128 for LDS).
template<typename P>
__device__ inline v16h make_a(P base, int g) {
    v8h lo = *(const v8h*)(base + (g ? 16 : 0));
    v8h hi = *(const v8h*)(base + (g ? 24 : 8));
    v16h a;
#pragma unroll
    for (int i = 0; i < 8; ++i) { a[i] = lo[i]; a[i + 8] = hi[i]; }
    return a;
}

__device__ inline v8f wmma16(v16h a, v16h b, v8f c) {
    return __builtin_amdgcn_wmma_f32_16x16x32_f16(false, a, false, b,
                                                  (short)0, c, false, false);
}

// Async-fill a 3x34xCIN f16 halo tile (replicate padding via clamped global addrs)
// into LDS starting at LDS byte offset 0, using CDNA5 async global->LDS copies
// (ASYNCcnt-tracked), then drain and barrier.
template<int CIN>
__device__ inline void async_fill_halo(const _Float16* __restrict__ in,
                                       int b, int h, int wbase) {
    constexpr int CHUNKS = 3 * 34 * CIN / 8;    // 16-byte chunks
    for (int i = threadIdx.x; i < CHUNKS; i += 256) {
        int cc = i % (CIN / 8);
        int p  = (i / (CIN / 8)) % 34;
        int r  = i / (CIN / 8) / 34;
        int hi_ = h + r - 1;     hi_ = hi_ < 0 ? 0 : (hi_ > HH - 1 ? HH - 1 : hi_);
        int wi_ = wbase + p - 1; wi_ = wi_ < 0 ? 0 : (wi_ > WW - 1 ? WW - 1 : wi_);
        const _Float16* gp = in + (((size_t)b * HH + hi_) * WW + wi_) * CIN + cc * 8;
        unsigned lds_off = (unsigned)(i * 16);
        asm volatile("global_load_async_to_lds_b128 %0, %1, off"
                     :: "v"(lds_off), "v"(gp) : "memory");
    }
    asm volatile("s_wait_asynccnt 0x0" ::: "memory");
    __syncthreads();
}

// ---------------- layout / packing kernels ----------------

__global__ __launch_bounds__(256) void pack_x_kernel(const float* __restrict__ x,
                                                     _Float16* __restrict__ xh) {
    size_t idx = (size_t)blockIdx.x * 256 + threadIdx.x;   // 4*64*256*256 total
    int c  = idx & 63;
    size_t p = idx >> 6;
    int w  = p & 255;
    int hh = (p >> 8) & 255;
    int bb = (int)(p >> 16);
    xh[idx] = (_Float16)x[(((size_t)bb * 64 + c) * HH + hh) * WW + w];
}

// w (256, Cin, KH, KW) f32 -> fragment-ordered f16 B tiles:
// wp[kk][cb][nt][lane][e] with B[k = e + 16*(lane>=16), n = lane&15].
__global__ __launch_bounds__(256) void pack_w_kernel(const float* __restrict__ w,
                                                     _Float16* __restrict__ wp,
                                                     int Cin, int KH, int KW) {
    int idx = blockIdx.x * 256 + threadIdx.x;
    int CB = Cin >> 5;
    int total = KH * KW * CB * 16 * 32 * 16;
    if (idx >= total) return;
    int e    = idx & 15;
    int lane = (idx >> 4) & 31;
    int nt   = (idx >> 9) & 15;
    int f    = idx >> 13;
    int cb   = f % CB;
    int kk   = f / CB;
    int kh   = kk / KW, kw = kk % KW;
    int g    = lane >> 4;
    int c    = cb * 32 + e + 16 * g;
    int o    = nt * 16 + (lane & 15);
    wp[idx] = (_Float16)w[(((size_t)o * Cin + c) * KH + kh) * KW + kw];
}

// mu (16,256) -> fragment-ordered 2*mu (N = clusters), plus mu2[k] = |mu_k|^2.
__global__ __launch_bounds__(256) void pack_mu_kernel(const float* __restrict__ mu,
                                                      _Float16* __restrict__ mup,
                                                      float* __restrict__ mu2) {
    int idx = blockIdx.x * 256 + threadIdx.x;
    if (idx < 4096) {
        int e    = idx & 15;
        int lane = (idx >> 4) & 31;
        int cb   = idx >> 9;
        int c    = cb * 32 + e + 16 * (lane >> 4);
        int k    = lane & 15;
        mup[idx] = (_Float16)(2.0f * mu[k * QQ + c]);
    }
    if (idx < 16) {
        float s = 0.f;
        for (int c = 0; c < QQ; ++c) { float v = mu[idx * QQ + c]; s += v * v; }
        mu2[idx] = s;
    }
}

// ---------------- conv1: 3x3 (64 -> 256), relu, q1 out ----------------
// Workgroup: 32 pixels x 256 out-ch; 8 waves, each 2 M-tiles x 2 N-tiles.
__global__ __launch_bounds__(256) void conv1_wmma_kernel(const _Float16* __restrict__ in,
                                                         const _Float16* __restrict__ wp,
                                                         const float* __restrict__ bias,
                                                         _Float16* __restrict__ out,
                                                         int guard) {
    constexpr int CIN = 64, CB = 2;
    __shared__ _Float16 halo[3 * 34 * CIN];      // async target at LDS offset 0
    // Opaque never-taken store: tells the compiler this LDS is written (the real
    // writes happen via the async asm), without constant-foldable contents.
    if (guard) halo[0] = (_Float16)(float)guard;

    int lane = threadIdx.x & 31;
    int wave = threadIdx.x >> 5;
    int g    = lane >> 4;
    int nl   = lane & 15;
    int tile = blockIdx.x;
    int wt = tile & 7;             // W/32 = 8 tiles
    int h  = (tile >> 3) & 255;
    int b  = tile >> 11;
    int wbase = wt * 32;
    int nt0 = wave * 2, nt1 = nt0 + 1;

    async_fill_halo<CIN>(in, b, h, wbase);

    v8f a00 = {}, a01 = {}, a10 = {}, a11 = {};
#pragma unroll
    for (int kk = 0; kk < 9; ++kk) {
        int kh = kk / 3, kw = kk % 3;
        const _Float16* r0 = &halo[((kh * 34) + (nl + kw)) * CIN];
        const _Float16* r1 = r0 + 16 * CIN;
#pragma unroll
        for (int cb = 0; cb < CB; ++cb) {
            v16h A0 = make_a(r0 + cb * 32, g);
            v16h A1 = make_a(r1 + cb * 32, g);
            const _Float16* wb = wp + ((size_t)(kk * CB + cb) * 512 + lane) * 16;
            __builtin_prefetch(wb + 8192, 0, 0);
            v16h B0 = *(const v16h*)(wb + (size_t)nt0 * 512);
            v16h B1 = *(const v16h*)(wb + (size_t)nt1 * 512);
            a00 = wmma16(A0, B0, a00); a01 = wmma16(A0, B1, a01);
            a10 = wmma16(A1, B0, a10); a11 = wmma16(A1, B1, a11);
        }
    }

    float bi0 = bias[nt0 * 16 + nl];
    float bi1 = bias[nt1 * 16 + nl];
    _Float16* ob = out + (((size_t)b * HH + h) * WW + wbase) * QQ;
#pragma unroll
    for (int v = 0; v < 8; ++v) {
        int m0 = v + 8 * g;
        float x00 = a00[v] + bi0; x00 = x00 > 0.f ? x00 : 0.f;
        float x01 = a01[v] + bi1; x01 = x01 > 0.f ? x01 : 0.f;
        float x10 = a10[v] + bi0; x10 = x10 > 0.f ? x10 : 0.f;
        float x11 = a11[v] + bi1; x11 = x11 > 0.f ? x11 : 0.f;
        ob[(size_t)m0 * QQ + nt0 * 16 + nl]        = (_Float16)x00;
        ob[(size_t)m0 * QQ + nt1 * 16 + nl]        = (_Float16)x01;
        ob[(size_t)(16 + m0) * QQ + nt0 * 16 + nl] = (_Float16)x10;
        ob[(size_t)(16 + m0) * QQ + nt1 * 16 + nl] = (_Float16)x11;
    }
}

// ---------------- fused: conv2 3x3 (256->256) + relu + 1x1 + cluster softmax ----------------
__global__ __launch_bounds__(256) void conv2_head_kernel(
    const _Float16* __restrict__ q1, const _Float16* __restrict__ w2p,
    const float* __restrict__ b2, const _Float16* __restrict__ w3p,
    const float* __restrict__ b3, const _Float16* __restrict__ mup,
    const float* __restrict__ mu2, const float* __restrict__ label,
    float* __restrict__ out, int guard) {
    constexpr int CIN = 256, CB = 8;
    __shared__ _Float16 smem[3 * 34 * CIN + 2 * 32 * 272];
    _Float16* halo = smem;                       // async target at LDS offset 0
    _Float16* q2t  = smem + 3 * 34 * CIN;
    _Float16* q3t  = q2t + 32 * 272;
    if (guard) smem[0] = (_Float16)(float)guard; // opaque never-taken store

    int lane = threadIdx.x & 31;
    int wave = threadIdx.x >> 5;
    int g    = lane >> 4;
    int nl   = lane & 15;
    int tile = blockIdx.x;
    int wt = tile & 7;
    int h  = (tile >> 3) & 255;
    int b  = tile >> 11;
    int wbase = wt * 32;
    int nt0 = wave * 2, nt1 = nt0 + 1;

    async_fill_halo<CIN>(q1, b, h, wbase);

    // ---- conv2 main loop: 72 k-steps x 4 WMMA ----
    v8f a00 = {}, a01 = {}, a10 = {}, a11 = {};
    for (int kk = 0; kk < 9; ++kk) {
        int kh = kk / 3, kw = kk % 3;
        const _Float16* r0 = &halo[((kh * 34) + (nl + kw)) * CIN];
        const _Float16* r1 = r0 + 16 * CIN;
#pragma unroll
        for (int cb = 0; cb < CB; ++cb) {
            v16h A0 = make_a(r0 + cb * 32, g);
            v16h A1 = make_a(r1 + cb * 32, g);
            const _Float16* wb = w2p + ((size_t)(kk * CB + cb) * 512 + lane) * 16;
            __builtin_prefetch(wb + 8192, 0, 0);
            v16h B0 = *(const v16h*)(wb + (size_t)nt0 * 512);
            v16h B1 = *(const v16h*)(wb + (size_t)nt1 * 512);
            a00 = wmma16(A0, B0, a00); a01 = wmma16(A0, B1, a01);
            a10 = wmma16(A1, B0, a10); a11 = wmma16(A1, B1, a11);
        }
    }
    {
        float bi0 = b2[nt0 * 16 + nl];
        float bi1 = b2[nt1 * 16 + nl];
#pragma unroll
        for (int v = 0; v < 8; ++v) {
            int m0 = v + 8 * g;
            float x00 = a00[v] + bi0; x00 = x00 > 0.f ? x00 : 0.f;
            float x01 = a01[v] + bi1; x01 = x01 > 0.f ? x01 : 0.f;
            float x10 = a10[v] + bi0; x10 = x10 > 0.f ? x10 : 0.f;
            float x11 = a11[v] + bi1; x11 = x11 > 0.f ? x11 : 0.f;
            q2t[m0 * 272 + nt0 * 16 + nl]        = (_Float16)x00;
            q2t[m0 * 272 + nt1 * 16 + nl]        = (_Float16)x01;
            q2t[(16 + m0) * 272 + nt0 * 16 + nl] = (_Float16)x10;
            q2t[(16 + m0) * 272 + nt1 * 16 + nl] = (_Float16)x11;
        }
    }
    __syncthreads();

    // ---- 1x1 conv: q3 = w3 * q2 + b3 (K=256) ----
    v8f c00 = {}, c01 = {}, c10 = {}, c11 = {};
#pragma unroll
    for (int cb = 0; cb < CB; ++cb) {
        v16h A0 = make_a(q2t + nl * 272 + cb * 32, g);
        v16h A1 = make_a(q2t + (16 + nl) * 272 + cb * 32, g);
        const _Float16* wb = w3p + ((size_t)cb * 512 + lane) * 16;
        v16h B0 = *(const v16h*)(wb + (size_t)nt0 * 512);
        v16h B1 = *(const v16h*)(wb + (size_t)nt1 * 512);
        c00 = wmma16(A0, B0, c00); c01 = wmma16(A0, B1, c01);
        c10 = wmma16(A1, B0, c10); c11 = wmma16(A1, B1, c11);
    }
    {
        float bi0 = b3[nt0 * 16 + nl];
        float bi1 = b3[nt1 * 16 + nl];
#pragma unroll
        for (int v = 0; v < 8; ++v) {
            int m0 = v + 8 * g;
            q3t[m0 * 272 + nt0 * 16 + nl]        = (_Float16)(c00[v] + bi0);
            q3t[m0 * 272 + nt1 * 16 + nl]        = (_Float16)(c01[v] + bi1);
            q3t[(16 + m0) * 272 + nt0 * 16 + nl] = (_Float16)(c10[v] + bi0);
            q3t[(16 + m0) * 272 + nt1 * 16 + nl] = (_Float16)(c11[v] + bi1);
        }
    }
    __syncthreads();

    // ---- cluster head: logit_k = 2 q.mu_k - |mu_k|^2 (q^2 cancels in softmax) ----
    if (wave < 2) {
        v8f acc;
#pragma unroll
        for (int v = 0; v < 8; ++v) acc[v] = -mu2[nl];    // lane -> cluster N
#pragma unroll
        for (int cb = 0; cb < CB; ++cb) {
            v16h A  = make_a(q3t + (wave * 16 + nl) * 272 + cb * 32, g);
            v16h Bm = *(const v16h*)(mup + ((size_t)cb * 32 + lane) * 16);
            acc = wmma16(A, Bm, acc);
        }
        float lab = label[nl];
        float* obase = out + ((size_t)b * HH + h) * WW + wbase + wave * 16;
#pragma unroll
        for (int v = 0; v < 8; ++v) {
            float t = acc[v];
            for (int off = 8; off; off >>= 1) t = fmaxf(t, __shfl_xor(t, off, 16));
            float e = __expf(acc[v] - t);
            float num = e * lab;
            float den = e;
            for (int off = 8; off; off >>= 1) {
                num += __shfl_xor(num, off, 16);
                den += __shfl_xor(den, off, 16);
            }
            if (nl == 0) obase[v + 8 * g] = num / den;
        }
    }
}

// ---------------- host-side launch ----------------
extern "C" void kernel_launch(void* const* d_in, const int* in_sizes, int n_in,
                              void* d_out, int out_size, void* d_ws, size_t ws_size,
                              hipStream_t stream) {
    (void)in_sizes; (void)n_in; (void)out_size; (void)ws_size;
    const float* x     = (const float*)d_in[0];
    const float* w1    = (const float*)d_in[1];
    const float* b1    = (const float*)d_in[2];
    const float* w2    = (const float*)d_in[3];
    const float* b2    = (const float*)d_in[4];
    const float* w3    = (const float*)d_in[5];
    const float* b3    = (const float*)d_in[6];
    const float* mu    = (const float*)d_in[7];
    const float* label = (const float*)d_in[8];
    float* out = (float*)d_out;

    char* ws = (char*)d_ws;
    size_t off = 0;
    auto alloc = [&](size_t bytes) -> void* {
        void* p = ws + off;
        off += (bytes + 255) & ~(size_t)255;
        return p;
    };
    _Float16* xh  = (_Float16*)alloc((size_t)4 * HH * WW * 64 * 2);   //  33.5 MB
    _Float16* q1  = (_Float16*)alloc((size_t)4 * HH * WW * QQ * 2);   // 134.2 MB
    _Float16* w1p = (_Float16*)alloc((size_t)9 * 64 * QQ * 2);
    _Float16* w2p = (_Float16*)alloc((size_t)9 * QQ * QQ * 2);
    _Float16* w3p = (_Float16*)alloc((size_t)QQ * QQ * 2);
    _Float16* mup = (_Float16*)alloc((size_t)4096 * 2);
    float*    mu2 = (float*)alloc(64);

    pack_x_kernel<<<(4 * HH * WW * 64) / 256, 256, 0, stream>>>(x, xh);
    pack_w_kernel<<<(9 * 64 * QQ + 255) / 256, 256, 0, stream>>>(w1, w1p, 64, 3, 3);
    pack_w_kernel<<<(9 * QQ * QQ + 255) / 256, 256, 0, stream>>>(w2, w2p, QQ, 3, 3);
    pack_w_kernel<<<(QQ * QQ + 255) / 256, 256, 0, stream>>>(w3, w3p, QQ, 1, 1);
    pack_mu_kernel<<<16, 256, 0, stream>>>(mu, mup, mu2);

    int tiles = 4 * HH * (WW / 32);   // 8192
    conv1_wmma_kernel<<<tiles, 256, 0, stream>>>(xh, w1p, b1, q1, 0);
    conv2_head_kernel<<<tiles, 256, 0, stream>>>(q1, w2p, b2, w3p, b3,
                                                 mup, mu2, label, out, 0);
}
